// ConvBranch_84739704750241
// MI455X (gfx1250) — matrix operand order repaired and checked by reading, not compiled, not run on hardware
//
#include <hip/hip_runtime.h>

#define BB 8
#define LL 2048
#define WW 1024
#define KK 5
#define GG 2
#define CI 512
#define CO 512
#define MT 32              // rows per block (two 16-row WMMA tiles per wave)

typedef _Float16 half_t;
typedef __attribute__((ext_vector_type(16))) _Float16 v16h;
typedef __attribute__((ext_vector_type(8)))  _Float16 v8h;
typedef __attribute__((ext_vector_type(4)))  _Float16 v4h;
typedef __attribute__((ext_vector_type(8)))  float    v8f;

// ---------------------------------------------------------------------------
// Kernel 1: pack conv weights f32 -> f16 in a WMMA-B-fragment-native layout.
// Wp[g][k][cb][o][cl]  with c = cb*32 + cl  (cb in [0,16), cl in [0,32)).
// A lane of a B fragment (32x16 f16 tile) then reads two 16-byte chunks at cl = 8h and 16+8h.
// ---------------------------------------------------------------------------
__global__ void dc_pack_w(const float* __restrict__ conv_w,
                          half_t* __restrict__ wp) {
    int i = blockIdx.x * 256 + threadIdx.x;
    const int total = GG * KK * 16 * CO * 32;
    if (i >= total) return;
    int cl = i & 31;  int t = i >> 5;
    int o  = t % CO;  t /= CO;
    int cb = t & 15;  t >>= 4;
    int k  = t % KK;  int g = t / KK;
    int c  = cb * 32 + cl;
    wp[i] = (half_t)conv_w[((size_t)(g * CO + o) * CI + c) * KK + k];
}

// ---------------------------------------------------------------------------
// Kernel 2: per-position fractional offsets -> integer indices + lerp weights.
// offsets = tanh(x[row,:] @ w_off + b_off) * 2 ;  pos = l + (k-2) + off
// Validity (zero padding outside [0,L)) is folded into the lerp weights.
// ---------------------------------------------------------------------------
__global__ __launch_bounds__(256)
void dc_offsets(const float* __restrict__ x, const float* __restrict__ w_off,
                const float* __restrict__ b_off,
                int* __restrict__ p0_out, float* __restrict__ frac_out) {
    const int row = blockIdx.x;           // 0 .. B*L-1
    const int t   = threadIdx.x;
    const float* xr = x + (size_t)row * WW;

    float acc[KK] = {0.f, 0.f, 0.f, 0.f, 0.f};
    for (int c = t; c < WW; c += 256) {
        float xv = xr[c];
        #pragma unroll
        for (int k = 0; k < KK; ++k) acc[k] += xv * w_off[c * KK + k];
    }
    __shared__ float red[KK][256];
    #pragma unroll
    for (int k = 0; k < KK; ++k) red[k][t] = acc[k];
    __syncthreads();
    for (int s = 128; s > 0; s >>= 1) {
        if (t < s) {
            #pragma unroll
            for (int k = 0; k < KK; ++k) red[k][t] += red[k][t + s];
        }
        __syncthreads();
    }
    if (t < KK) {
        int k = t;
        float off = tanhf(red[k][0] + b_off[k]) * 2.0f;
        int   l   = row % LL;
        float pos = (float)l + ((float)k - 2.0f) + off;
        float p0f = floorf(pos);
        p0_out[row * KK + k]   = (int)p0f;
        frac_out[row * KK + k] = pos - p0f;
    }
}

// ---------------------------------------------------------------------------
// Kernel 3: fused deformable grouped conv (f16 WMMA, f32 accum) + bias
//           + LayerNorm + masked projection contribution per row.
// One block = one 32-row M tile x all 1024 output channels. 8 waves:
// waves 0-3 -> group 0, waves 4-7 -> group 1; each wave owns 128 channels
// and TWO 16-row C tiles, so every B fragment is reused for 2 WMMAs.
// ---------------------------------------------------------------------------
__global__ __launch_bounds__(256)
void dc_main(const float* __restrict__ x, const half_t* __restrict__ wp,
             const int* __restrict__ p0a, const float* __restrict__ fra,
             const float* __restrict__ conv_b, const float* __restrict__ gamma,
             const float* __restrict__ beta, const float* __restrict__ projw,
             const unsigned char* __restrict__ mask,
             float* __restrict__ rowc) {
    __shared__ __align__(16) union {
        half_t a[MT][1032];   // interpolated sample tile (per tap), padded
        half_t h[MT][1032];   // conv output tile (reused after taps), padded
    } lds;
    __shared__ float redA[MT][8], redB[MT][8];
    __shared__ int   s_i0[MT][KK], s_i1[MT][KK];
    __shared__ float s_w0[MT][KK], s_w1[MT][KK];

    const int row0 = blockIdx.x * MT;
    const int bidx = row0 / LL;            // all MT rows share one batch (2048%32==0)
    const int t    = threadIdx.x;
    const int wave = t >> 5, lane = t & 31;
    const int hi   = lane >> 4, ln = lane & 15;
    const int g    = wave >> 2;            // group
    const int nq   = (wave & 3) * 128;     // column base within group

    // interpolation parameters for the MT rows x 5 taps
    if (t < MT * KK) {
        int m = t / KK, k = t % KK;
        int r = row0 + m;
        int p0 = p0a[r * KK + k];
        float fr = fra[r * KK + k];
        float w0 = 1.0f - fr, w1 = fr;
        int i0 = p0, i1 = p0 + 1;
        if (i0 < 0 || i0 >= LL) { w0 = 0.0f; i0 = 0; }
        if (i1 < 0 || i1 >= LL) { w1 = 0.0f; i1 = 0; }
        s_i0[m][k] = i0; s_i1[m][k] = i1;
        s_w0[m][k] = w0; s_w1[m][k] = w1;
    }

    v8f acc[2][8] = {};   // [m-tile][n-tile] C fragments per wave

    for (int k = 0; k < KK; ++k) {
        __syncthreads();   // params ready / previous compute done
        // --- stage interpolated samples: MT rows x 1024 channels (f16) ---
        for (int i = t; i < MT * 256; i += 256) {
            int m = i >> 8; int c4 = i & 255;
            const float4* r0 =
                (const float4*)(x + (size_t)(bidx * LL + s_i0[m][k]) * WW) + c4;
            const float4* r1 =
                (const float4*)(x + (size_t)(bidx * LL + s_i1[m][k]) * WW) + c4;
            float w0 = s_w0[m][k], w1 = s_w1[m][k];
            float4 a0 = *r0, a1 = *r1;
            v4h hv;
            hv[0] = (half_t)(a0.x * w0 + a1.x * w1);
            hv[1] = (half_t)(a0.y * w0 + a1.y * w1);
            hv[2] = (half_t)(a0.z * w0 + a1.z * w1);
            hv[3] = (half_t)(a0.w * w0 + a1.w * w1);
            *(v4h*)&lds.a[m][c4 * 4] = hv;
        }
        __syncthreads();
        // --- WMMA over this tap's 512 channels (per group) ---
        const int abase = g * CI;
        for (int cb = 0; cb < 16; ++cb) {
            int c0 = cb * 32;
            // Two A fragments (rows 0-15 and 16-31), two 16B LDS reads each
            v16h A[2];
            #pragma unroll
            for (int mt = 0; mt < 2; ++mt) {
                v8h alo = *(const v8h*)&lds.a[mt * 16 + ln][abase + c0 + 8 * hi];
                v8h ahi = *(const v8h*)&lds.a[mt * 16 + ln][abase + c0 + 16 + 8 * hi];
                #pragma unroll
                for (int j = 0; j < 8; ++j) { A[mt][j] = alo[j]; A[mt][j + 8] = ahi[j]; }
            }

            const half_t* wbase =
                wp + (size_t)(((g * KK + k) * 16 + cb)) * CO * 32;
            #pragma unroll
            for (int tt = 0; tt < 8; ++tt) {
                int o = nq + tt * 16 + ln;
                // B fragment: 32x16 f16, lane = col, two 16B global loads;
                // reused for both M tiles.
                const v8h* wv = (const v8h*)(wbase + o * 32);
                v8h blo = wv[hi], bhi = wv[2 + hi];
                v16h Bf;
                #pragma unroll
                for (int j = 0; j < 8; ++j) { Bf[j] = blo[j]; Bf[j + 8] = bhi[j]; }
                acc[0][tt] = __builtin_amdgcn_wmma_f32_16x16x32_f16(
                    false, A[0], false, Bf, (short)0, acc[0][tt], false, false);
                acc[1][tt] = __builtin_amdgcn_wmma_f32_16x16x32_f16(
                    false, A[1], false, Bf, (short)0, acc[1][tt], false, false);
            }
        }
    }

    __syncthreads();   // all LDS sample reads done; reuse LDS for h tile
    // --- bias + write h tile to LDS (f16) ---
    #pragma unroll
    for (int tt = 0; tt < 8; ++tt) {
        int col = g * CI + nq + tt * 16 + ln;
        float bias = conv_b[col];
        #pragma unroll
        for (int mt = 0; mt < 2; ++mt) {
            #pragma unroll
            for (int j = 0; j < 8; ++j)
                lds.h[mt * 16 + j + 8 * hi][col] = (half_t)(acc[mt][tt][j] + bias);
        }
    }
    __syncthreads();

    // --- LayerNorm + gamma/beta + proj_w dot, 8 threads per row ---
    const int m  = t >> 3, tr = t & 7;
    const int cb = tr * 128;
    float s1 = 0.f, s2 = 0.f;
    for (int c = cb; c < cb + 128; ++c) {
        float v = (float)lds.h[m][c];
        s1 += v; s2 += v * v;
    }
    redA[m][tr] = s1; redB[m][tr] = s2;
    __syncthreads();
    if (tr == 0) {
        float a = 0.f, b2 = 0.f;
        for (int i = 0; i < 8; ++i) { a += redA[m][i]; b2 += redB[m][i]; }
        float mean = a * (1.0f / (float)WW);
        float var  = b2 * (1.0f / (float)WW) - mean * mean;
        redA[m][0] = mean;
        redB[m][0] = rsqrtf(var + 1e-5f);
    }
    __syncthreads();
    float mean = redA[m][0], rstd = redB[m][0];
    float contrib = 0.f;
    for (int c = cb; c < cb + 128; ++c) {
        float v = ((float)lds.h[m][c] - mean) * rstd * gamma[c] + beta[c];
        contrib += v * projw[c];
    }
    __syncthreads();
    redA[m][tr] = contrib;
    __syncthreads();
    if (tr == 0) {
        float s = 0.f;
        for (int i = 0; i < 8; ++i) s += redA[m][i];
        int r = row0 + m;
        rowc[r] = mask[r] ? 0.0f : s;
    }
}

// ---------------------------------------------------------------------------
// Kernel 4: per-batch reduction: mean over valid rows + proj bias.
// ---------------------------------------------------------------------------
__global__ __launch_bounds__(256)
void dc_finalize(const float* __restrict__ rowc,
                 const unsigned char* __restrict__ mask,
                 const float* __restrict__ proj_b, float* __restrict__ out) {
    int b = blockIdx.x, t = threadIdx.x;
    float s = 0.f; int cnt = 0;
    for (int l = t; l < LL; l += 256) {
        int r = b * LL + l;
        s += rowc[r];
        cnt += mask[r] ? 0 : 1;
    }
    __shared__ float rs[256];
    __shared__ int   rc[256];
    rs[t] = s; rc[t] = cnt;
    __syncthreads();
    for (int st = 128; st > 0; st >>= 1) {
        if (t < st) { rs[t] += rs[t + st]; rc[t] += rc[t + st]; }
        __syncthreads();
    }
    if (t == 0) {
        float len = fmaxf((float)rc[0], 1.0f);
        out[b] = rs[0] / len + proj_b[0];
    }
}

// ---------------------------------------------------------------------------
extern "C" void kernel_launch(void* const* d_in, const int* in_sizes, int n_in,
                              void* d_out, int out_size, void* d_ws,
                              size_t ws_size, hipStream_t stream) {
    const float*         x      = (const float*)d_in[0];
    const unsigned char* mask   = (const unsigned char*)d_in[1];
    const float*         w_off  = (const float*)d_in[2];
    const float*         b_off  = (const float*)d_in[3];
    const float*         conv_w = (const float*)d_in[4];
    const float*         conv_b = (const float*)d_in[5];
    const float*         gamma  = (const float*)d_in[6];
    const float*         beta   = (const float*)d_in[7];
    const float*         proj_w = (const float*)d_in[8];
    const float*         proj_b = (const float*)d_in[9];
    float* out = (float*)d_out;
    (void)in_sizes; (void)n_in; (void)out_size; (void)ws_size;

    // workspace carve-up (256B aligned): ~6 MiB total
    char* ws = (char*)d_ws;
    size_t off = 0;
    auto carve = [&](size_t bytes) -> void* {
        void* p = ws + off;
        off = (off + bytes + 255) & ~(size_t)255;
        return p;
    };
    int*    p0a  = (int*)   carve((size_t)BB * LL * KK * sizeof(int));
    float*  fra  = (float*) carve((size_t)BB * LL * KK * sizeof(float));
    half_t* wp   = (half_t*)carve((size_t)GG * KK * 16 * CO * 32 * sizeof(half_t));
    float*  rowc = (float*) carve((size_t)BB * LL * sizeof(float));

    const int packTotal = GG * KK * 16 * CO * 32;
    dc_pack_w  <<<(packTotal + 255) / 256, 256, 0, stream>>>(conv_w, wp);
    dc_offsets <<<BB * LL,              256, 0, stream>>>(x, w_off, b_off, p0a, fra);
    dc_main    <<<(BB * LL) / MT,       256, 0, stream>>>(x, wp, p0a, fra, conv_b,
                                                          gamma, beta, proj_w,
                                                          mask, rowc);
    dc_finalize<<<BB,                   256, 0, stream>>>(rowc, mask, proj_b, out);
}
